// MultiHeadAttetion_35768487641222
// MI455X (gfx1250) — matrix operand order repaired and checked by reading, not compiled, not run on hardware
//
#include <hip/hip_runtime.h>

// ---------------------------------------------------------------------------
// MI455X (gfx1250) multi-head causal attention, f16 WMMA + f32 accumulate.
// Pipeline: cvt(x), transpose-cvt(W*), GEMM(Q), GEMM(K), GEMM(V->transposed),
//           flash-attention (online softmax, per-wave causal trip count),
//           GEMM(out, f32).
// GEMM k-loop is double-buffered through LDS using CDNA5 async-to-LDS loads
// (ASYNCcnt) for the A tile and the Tensor Data Mover (TENSORcnt) for the
// B tile, with plain-load fallbacks guarded by __has_builtin.
// ---------------------------------------------------------------------------

typedef _Float16 v8h  __attribute__((ext_vector_type(8)));
typedef _Float16 v16h __attribute__((ext_vector_type(16)));
typedef float    v8f  __attribute__((ext_vector_type(8)));

#define B_  2
#define S_  2048
#define D_  1024
#define H_  16
#define HD_ 64

#define HAVE_ASYNC_LDS __has_builtin(__builtin_amdgcn_global_load_async_to_lds_b128)
#define HAVE_TDM       __has_builtin(__builtin_amdgcn_tensor_load_to_lds)

#if __has_builtin(__builtin_amdgcn_s_wait_asynccnt)
#define WAIT_ASYNC(n) __builtin_amdgcn_s_wait_asynccnt(n)
#else
#define WAIT_ASYNC(n) asm volatile("s_wait_asynccnt " #n ::: "memory")
#endif

#if __has_builtin(__builtin_amdgcn_s_wait_tensorcnt)
#define WAIT_TENSOR(n) __builtin_amdgcn_s_wait_tensorcnt(n)
#else
#define WAIT_TENSOR(n) asm volatile("s_wait_tensorcnt " #n ::: "memory")
#endif

// Async-to-LDS b128: builtin takes v4i pointers in AS(1)/AS(3).
typedef int asy_v4i __attribute__((vector_size(16)));
typedef __attribute__((address_space(1))) asy_v4i as1_v4i;
typedef __attribute__((address_space(3))) asy_v4i as3_v4i;

__device__ __forceinline__ void async_cp_b128(const void* gp, void* lp) {
#if HAVE_ASYNC_LDS
  __builtin_amdgcn_global_load_async_to_lds_b128((as1_v4i*)gp, (as3_v4i*)lp,
                                                 0, 0);
#else
  *(v8h*)lp = *(const v8h*)gp;
#endif
}

// Build a 16-wide f16 A/B fragment from two contiguous 16B chunks.
__device__ __forceinline__ v16h ld_frag2(const _Float16* p0, const _Float16* p1) {
  v8h lo = *(const v8h*)p0;
  v8h hi = *(const v8h*)p1;
  return __builtin_shufflevector(lo, hi, 0,1,2,3,4,5,6,7,8,9,10,11,12,13,14,15);
}

__device__ __forceinline__ v8f wmma16(v16h a, v16h b, v8f c) {
  // D = A(16x32 f16) * B(32x16 f16) + C(16x16 f32)
  return __builtin_amdgcn_wmma_f32_16x16x32_f16(false, a, false, b, (short)0, c,
                                                false, false);
}

__device__ __forceinline__ void wait_lds() {
  asm volatile("s_wait_dscnt 0" ::: "memory");
}

__device__ __forceinline__ float xor_max16(float v) {
#pragma unroll
  for (int m = 1; m < 16; m <<= 1) v = fmaxf(v, __shfl_xor(v, m, 32));
  return v;
}
__device__ __forceinline__ float xor_sum16(float v) {
#pragma unroll
  for (int m = 1; m < 16; m <<= 1) v += __shfl_xor(v, m, 32);
  return v;
}

#if HAVE_TDM
typedef unsigned int u32x4 __attribute__((ext_vector_type(4)));
typedef int          i32x4 __attribute__((ext_vector_type(4)));
typedef int          i32x8 __attribute__((ext_vector_type(8)));

// 2D TDM load: tile (tile_d0 contiguous elems) x (tile_d1 lines) of f16,
// line stride stride0 (elements), packed linearly into LDS at lds_off.
// D# field packing per CDNA5 ISA ch.8 (group0: count/lds/global/type,
// group1: data_size=2B, tensor dims, tile dims, dim0 stride).
__device__ __forceinline__ void tdm_load_2d_f16(unsigned lds_off,
                                                const _Float16* gptr,
                                                int tensor_d0, int tensor_d1,
                                                int tile_d0, int tile_d1,
                                                int stride0) {
  unsigned long long ga = (unsigned long long)gptr;
  u32x4 g0;
  g0[0] = 1u;                                   // count=1, user descriptor
  g0[1] = lds_off;                              // LDS byte address
  g0[2] = (unsigned)(ga & 0xffffffffu);         // global_addr[31:0]
  g0[3] = (unsigned)((ga >> 32) & 0x1ffffffu)   // global_addr[56:32]
          | (2u << 30);                         // type = 2 ("image")
  i32x8 g1;
  g1[0] = 1 << 16;                              // data_size=1 -> 2 bytes
  g1[1] = (tensor_d0 & 0xffff) << 16;           // tensor_dim0[15:0]
  g1[2] = ((tensor_d0 >> 16) & 0xffff) | ((tensor_d1 & 0xffff) << 16);
  g1[3] = ((tensor_d1 >> 16) & 0xffff) | ((tile_d0 & 0xffff) << 16);
  g1[4] = tile_d1 & 0xffff;                     // tile_dim1 (tile_dim2 = 0)
  g1[5] = stride0;                              // tensor_dim0_stride[31:0]
  g1[6] = 0;
  g1[7] = 0;
  i32x4 z4 = {0, 0, 0, 0};
#if __clang_major__ >= 23
  i32x8 z8 = {0, 0, 0, 0, 0, 0, 0, 0};
  __builtin_amdgcn_tensor_load_to_lds(g0, g1, z4, z4, z8, 0);
#else
  __builtin_amdgcn_tensor_load_to_lds(g0, g1, z4, z4, 0);
#endif
}
#endif  // HAVE_TDM

// ---------------------------------------------------------------------------
// Conversion kernels
// ---------------------------------------------------------------------------
__global__ __launch_bounds__(256) void cvt_f32_to_f16_kernel(
    const float* __restrict__ in, _Float16* __restrict__ out, int n) {
  int i = blockIdx.x * 256 + threadIdx.x;
  if (i < n) out[i] = (_Float16)in[i];
}

// W [K=1024][N=1024] f32 row-major  ->  WT [N][K] f16
__global__ __launch_bounds__(256) void cvt_transpose_kernel(
    const float* __restrict__ W, _Float16* __restrict__ WT) {
  int i = blockIdx.x * 256 + threadIdx.x;  // over 1024*1024
  int k = i >> 10;
  int n = i & 1023;
  WT[(size_t)n * D_ + k] = (_Float16)W[i];
}

// ---------------------------------------------------------------------------
// GEMM: C[M,N] = A[M,K] * W[K,N] + bias, A f16 [M][K], WT f16 [N][K].
// Block: 256 thr = 8 waves, block tile 128x128, wave tile 32x64 (2x4 subtiles).
// Double-buffered LDS staging: A via async-to-LDS, B via TDM (wave 0).
// MODE 0: f16 out, row-major [M][N]         (Q, K projections)
// MODE 1: f16 out, transposed [B][H][HD][S] (V projection, for P*V B-operand)
// MODE 2: f32 out, row-major                (output projection)
// ---------------------------------------------------------------------------
template <int MODE>
__global__ __launch_bounds__(256) void gemm_wmma_kernel(
    const _Float16* __restrict__ A, const _Float16* __restrict__ WT,
    const float* __restrict__ bias, void* __restrict__ Cout) {
  __shared__ alignas(128) _Float16 sA[2][128 * 32];  // [row][k] 8KB each
  __shared__ alignas(128) _Float16 sB[2][128 * 32];  // [col][k] 8KB each

  const int lane = threadIdx.x & 31;
  const int wave = threadIdx.x >> 5;
  const int half = lane >> 4;   // lane group 0/1
  const int l16  = lane & 15;
  const int wm = wave >> 1;     // 0..3  -> 32-row slice
  const int wn = wave & 1;      // 0..1  -> 64-col slice
  const int row0 = blockIdx.y * 128;
  const int col0 = blockIdx.x * 128;

  // Stage the 128x32 A tile and the 128x32 (transposed-W) B tile for k=kk.
  auto issue_tile = [&](int kk, int buf) {
    // ---- A tile: 2 async b128 per wave (each instr moves 32 lanes * 16B) --
#pragma unroll
    for (int q = 0; q < 2; ++q) {
      const int r = wave * 16 + q * 8 + (lane >> 2);  // 0..127
      const int c = (lane & 3) * 8;                   // 0/8/16/24
      async_cp_b128(A + (size_t)(row0 + r) * D_ + kk + c,
                    &sA[buf][r * 32 + c]);
    }
    // ---- B tile -----------------------------------------------------------
#if HAVE_TDM
    if (wave == 0) {
      tdm_load_2d_f16((unsigned)(unsigned long long)&sB[buf][0],
                      WT + (size_t)col0 * D_ + kk,
                      /*tensor_d0=*/D_, /*tensor_d1=*/D_,
                      /*tile_d0=*/32, /*tile_d1=*/128, /*stride0=*/D_);
    }
#else
#pragma unroll
    for (int q = 0; q < 2; ++q) {
      const int r = wave * 16 + q * 8 + (lane >> 2);
      const int c = (lane & 3) * 8;
      async_cp_b128(WT + (size_t)(col0 + r) * D_ + kk + c,
                    &sB[buf][r * 32 + c]);
    }
#endif
  };

  v8f acc[2][4];
#pragma unroll
  for (int mi = 0; mi < 2; ++mi)
#pragma unroll
    for (int ni = 0; ni < 4; ++ni) { v8f z = {}; acc[mi][ni] = z; }

  const int NK = D_ / 32;  // 32 k-steps
  issue_tile(0, 0);

#pragma unroll 1
  for (int i = 0; i < NK; ++i) {
    const int cur = i & 1;
    if (i + 1 < NK) {
      issue_tile((i + 1) * 32, cur ^ 1);
      // async/tensor ops complete in order: waiting for <= (newer batch size)
      // guarantees tile i is fully in LDS.
#if HAVE_ASYNC_LDS
      WAIT_ASYNC(2);
#endif
#if HAVE_TDM
      if (wave == 0) WAIT_TENSOR(1);
#endif
    } else {
#if HAVE_ASYNC_LDS
      WAIT_ASYNC(0);
#endif
#if HAVE_TDM
      if (wave == 0) WAIT_TENSOR(0);
#endif
    }
    __syncthreads();

    const _Float16* sac = sA[cur];
    const _Float16* sbc = sB[cur];
    v16h af[2];
#pragma unroll
    for (int mi = 0; mi < 2; ++mi) {
      const _Float16* ap = sac + (wm * 32 + mi * 16 + l16) * 32 + half * 16;
      af[mi] = ld_frag2(ap, ap + 8);
    }
    v16h bf[4];
#pragma unroll
    for (int ni = 0; ni < 4; ++ni) {
      const _Float16* bp = sbc + (wn * 64 + ni * 16 + l16) * 32 + half * 16;
      bf[ni] = *(const v16h*)bp;  // 32B contiguous in LDS
    }
#pragma unroll
    for (int mi = 0; mi < 2; ++mi)
#pragma unroll
      for (int ni = 0; ni < 4; ++ni)
        acc[mi][ni] = wmma16(af[mi], bf[ni], acc[mi][ni]);

    __syncthreads();  // protect buffers before next-next issue overwrites
  }

#pragma unroll
  for (int ni = 0; ni < 4; ++ni) {
    const int c = col0 + wn * 64 + ni * 16 + l16;
    const float bv = bias[c];
#pragma unroll
    for (int mi = 0; mi < 2; ++mi) {
#pragma unroll
      for (int r = 0; r < 8; ++r) {
        const int g = row0 + wm * 32 + mi * 16 + 8 * half + r;  // b*S + s
        const float val = acc[mi][ni][r] + bv;
        if (MODE == 0) {
          ((_Float16*)Cout)[(size_t)g * D_ + c] = (_Float16)val;
        } else if (MODE == 1) {
          const int b  = g >> 11, s = g & (S_ - 1);
          const int hh = c >> 6,  hd = c & (HD_ - 1);
          ((_Float16*)Cout)[((size_t)((b * H_ + hh) * HD_ + hd) << 11) + s] =
              (_Float16)val;
        } else {
          ((float*)Cout)[(size_t)g * D_ + c] = val;
        }
      }
    }
  }
}

// ---------------------------------------------------------------------------
// Flash attention. Grid: (S/128, B*H), 256 threads = 8 waves.
// Each wave owns 16 query rows; iterates its own causal key range in blocks
// of 32 keys. QK^T and P*V both via 16x16x32 f16 WMMA; P goes through a
// 1KB/wave LDS buffer to convert C-layout -> A-layout (s_wait_dscnt sync).
// ---------------------------------------------------------------------------
__global__ __launch_bounds__(256) void attn_fa_kernel(
    const _Float16* __restrict__ Q, const _Float16* __restrict__ K,
    const _Float16* __restrict__ VT, _Float16* __restrict__ ctx) {
  __shared__ alignas(64) _Float16 pbuf[8][16 * 32];

  const int lane = threadIdx.x & 31;
  const int wave = threadIdx.x >> 5;
  const int half = lane >> 4;
  const int l16  = lane & 15;
  const int qtile = blockIdx.x * 128 + wave * 16;
  const int b = blockIdx.y >> 4;
  const int h = blockIdx.y & (H_ - 1);

  // Q fragments (16 rows x 64), kept in registers for the whole loop.
  const _Float16* qrow = Q + (size_t)(b * S_ + qtile + l16) * D_ + h * HD_;
  v16h qf[2];
#pragma unroll
  for (int hc = 0; hc < 2; ++hc) {
    const _Float16* p = qrow + hc * 32 + half * 16;
    qf[hc] = ld_frag2(p, p + 8);
  }

  v8f O[4];
#pragma unroll
  for (int t = 0; t < 4; ++t) { v8f z = {}; O[t] = z; }
  float mst[8], lst[8];
#pragma unroll
  for (int r = 0; r < 8; ++r) { mst[r] = -__builtin_inff(); lst[r] = 0.f; }

  const _Float16* kbh = K + (size_t)(b * S_) * D_ + h * HD_;
  const _Float16* vbh = VT + (size_t)(b * H_ + h) * HD_ * S_;  // [HD][S]
  _Float16* pl = pbuf[wave];

  const int nkb = (qtile + 16 + 31) >> 5;  // causal: keys <= qtile+15 needed
  for (int kb = 0; kb < nkb; ++kb) {
    const int j = kb * 32;

    // prefetch next key/value blocks into cache while we compute this one
    __builtin_prefetch(kbh + (size_t)(j + 32) * D_, 0, 3);
    __builtin_prefetch(vbh + (size_t)l16 * S_ + j + 32, 0, 3);

    // scores = Q * K^T  (two 16x16 tiles covering keys j..j+31)
    v8f sc[2];
#pragma unroll
    for (int nt = 0; nt < 2; ++nt) {
      v8f a = {};
#pragma unroll
      for (int hc = 0; hc < 2; ++hc) {
        const _Float16* kp =
            kbh + (size_t)(j + nt * 16 + l16) * D_ + hc * 32 + half * 16;
        a = wmma16(qf[hc], *(const v16h*)kp, a);  // B = K^T, contiguous 32B
      }
      sc[nt] = a;
    }

    // online softmax (scale 1/sqrt(64)=0.125, causal mask -1e10)
#pragma unroll
    for (int r = 0; r < 8; ++r) {
      const int qidx = qtile + 8 * half + r;
      float s0 = (j + l16      <= qidx) ? sc[0][r] * 0.125f : -1.0e10f;
      float s1 = (j + 16 + l16 <= qidx) ? sc[1][r] * 0.125f : -1.0e10f;
      float mx = xor_max16(fmaxf(s0, s1));
      float mnew = fmaxf(mst[r], mx);
      float corr = __expf(mst[r] - mnew);
      float p0 = __expf(s0 - mnew);
      float p1 = __expf(s1 - mnew);
      float rs = xor_sum16(p0 + p1);
      lst[r] = lst[r] * corr + rs;
      mst[r] = mnew;
#pragma unroll
      for (int t = 0; t < 4; ++t) O[t][r] *= corr;
      const int m = 8 * half + r;
      pl[m * 32 + l16]      = (_Float16)p0;  // P in [row][key] LDS layout
      pl[m * 32 + 16 + l16] = (_Float16)p1;
    }
    wait_lds();

    // P (16x32) as A-fragment, V^T rows as B-fragments (contiguous 32B).
    const _Float16* pp = pl + l16 * 32 + half * 16;
    v16h pf = ld_frag2(pp, pp + 8);
#pragma unroll
    for (int t = 0; t < 4; ++t) {
      const _Float16* vp = vbh + (size_t)(t * 16 + l16) * S_ + j + half * 16;
      O[t] = wmma16(pf, *(const v16h*)vp, O[t]);
    }
  }

  // normalize and store context in [B*S][D] f16 (input of output projection)
#pragma unroll
  for (int r = 0; r < 8; ++r) {
    const float inv = 1.0f / lst[r];
    const size_t row = (size_t)(b * S_ + qtile + 8 * half + r);
#pragma unroll
    for (int t = 0; t < 4; ++t)
      ctx[row * D_ + h * HD_ + t * 16 + l16] = (_Float16)(O[t][r] * inv);
  }
}

// ---------------------------------------------------------------------------
// Host-side launch
// ---------------------------------------------------------------------------
extern "C" void kernel_launch(void* const* d_in, const int* in_sizes, int n_in,
                              void* d_out, int out_size, void* d_ws,
                              size_t ws_size, hipStream_t stream) {
  (void)in_sizes; (void)n_in; (void)out_size; (void)ws_size;

  const float* x  = (const float*)d_in[0];
  const float* Wq = (const float*)d_in[1];
  const float* bq = (const float*)d_in[2];
  const float* Wk = (const float*)d_in[3];
  const float* bk = (const float*)d_in[4];
  const float* Wv = (const float*)d_in[5];
  const float* bv = (const float*)d_in[6];
  const float* Wo = (const float*)d_in[7];
  const float* bo = (const float*)d_in[8];

  char* ws = (char*)d_ws;
  const size_t MB = (size_t)1 << 20;
  _Float16* x16  = (_Float16*)(ws + 0 * MB);   // 8 MiB  [4096][1024]
  _Float16* WqT  = (_Float16*)(ws + 8 * MB);   // 2 MiB  [N][K]
  _Float16* WkT  = (_Float16*)(ws + 10 * MB);
  _Float16* WvT  = (_Float16*)(ws + 12 * MB);
  _Float16* WoT  = (_Float16*)(ws + 14 * MB);
  _Float16* Q16  = (_Float16*)(ws + 16 * MB);  // 8 MiB  [4096][1024]
  _Float16* K16  = (_Float16*)(ws + 24 * MB);  // 8 MiB  [4096][1024]
  _Float16* VT16 = (_Float16*)(ws + 32 * MB);  // 8 MiB  [B][H][HD][S]
  _Float16* C16  = (_Float16*)(ws + 40 * MB);  // 8 MiB  [4096][1024]

  const int nx = B_ * S_ * D_;  // 4M elements
  cvt_f32_to_f16_kernel<<<nx / 256, 256, 0, stream>>>(x, x16, nx);
  cvt_transpose_kernel<<<(D_ * D_) / 256, 256, 0, stream>>>(Wq, WqT);
  cvt_transpose_kernel<<<(D_ * D_) / 256, 256, 0, stream>>>(Wk, WkT);
  cvt_transpose_kernel<<<(D_ * D_) / 256, 256, 0, stream>>>(Wv, WvT);
  cvt_transpose_kernel<<<(D_ * D_) / 256, 256, 0, stream>>>(Wo, WoT);

  dim3 ggrid(D_ / 128, (B_ * S_) / 128);  // (8, 32)
  gemm_wmma_kernel<0><<<ggrid, 256, 0, stream>>>(x16, WqT, bq, (void*)Q16);
  gemm_wmma_kernel<0><<<ggrid, 256, 0, stream>>>(x16, WkT, bk, (void*)K16);
  gemm_wmma_kernel<1><<<ggrid, 256, 0, stream>>>(x16, WvT, bv, (void*)VT16);

  attn_fa_kernel<<<dim3(S_ / 128, B_ * H_), 256, 0, stream>>>(Q16, K16, VT16,
                                                              C16);

  gemm_wmma_kernel<2><<<ggrid, 256, 0, stream>>>(C16, WoT, bo, d_out);
}
